// TrajectoryLSTM_53566832116174
// MI455X (gfx1250) — matrix operand order repaired and checked by reading, not compiled, not run on hardware
//
#include <hip/hip_runtime.h>

typedef __attribute__((ext_vector_type(16))) __bf16 v16bf;
typedef __attribute__((ext_vector_type(8)))  float  v8f;
typedef __attribute__((ext_vector_type(4)))  unsigned int v4u;
typedef __attribute__((ext_vector_type(8)))  int    v8i;
typedef __attribute__((ext_vector_type(4)))  int    v4i;

__device__ __forceinline__ unsigned short f2bf(float f) {
  unsigned int u = __builtin_bit_cast(unsigned int, f);
  u += 0x7FFFu + ((u >> 16) & 1u);       // RNE, for prep-side math
  return (unsigned short)(u >> 16);
}

// native f32 -> bf16 (v_cvt), 1 VALU op
__device__ __forceinline__ unsigned short f2bf_n(float f) {
  return __builtin_bit_cast(unsigned short, (__bf16)f);
}

// hardware tanh (V_TANH_F32 on gfx1250), else exp fallback
__device__ __forceinline__ float hw_tanh(float x) {
#if __has_builtin(__builtin_amdgcn_tanhf)
  return __builtin_amdgcn_tanhf(x);
#elif __has_builtin(__builtin_amdgcn_tanh_f32)
  return __builtin_amdgcn_tanh_f32(x);
#else
  float e = __expf(2.0f * x);
  return 1.0f - 2.0f * __frcp_rn(e + 1.0f);
#endif
}

// sigmoid(x + b) with hb = 0.5*b pre-scaled: fma, tanh, fma
__device__ __forceinline__ float sgate(float x, float hb) {
  return __builtin_fmaf(0.5f, hw_tanh(__builtin_fmaf(0.5f, x, hb)), 0.5f);
}

// WMMA-A staging address math: element (m, K) of a 16-bit A matrix lives in
// lane m + 16*((K%32)>>3 & 1), half index (K&7)|((K&16)>>1), chunk K/32.
// Staging is stored frag-contiguous: u16 index = (chunk*32 + lane)*16 + half.
__device__ __forceinline__ int fragIdx(int K) {
  int c  = K >> 5, kr = K & 31;
  int lo = ((kr >> 3) & 1) << 4;
  int i  = (kr & 7) | ((kr & 16) >> 1);
  return ((c * 32 + lo) << 4) + i;      // add m*16 for row m
}

// ---------------------------------------------------------------------------
// Prep: pack combined recurrent weights into per-lane WMMA-B fragment layout.
// B0 (layer0): rows 0..3 = W_ih0^T, rows 4..67 = W_hh0^T, rows 68..95 = 0
// B1 (layer1): rows 0..63 = W_ih1^T, rows 64..127 = W_hh1^T
// Lane l, half i of frag (tile,chunk): K = chunk*32 + (i&7) + ((i&8)<<1) + (l>=16?8:0),
// N = tile*16 + (l&15); 16 halves contiguous -> 2x ds_load_b128 at use.
// ---------------------------------------------------------------------------
__global__ void prep_weights(const float* __restrict__ Wih0,
                             const float* __restrict__ Whh0,
                             const float* __restrict__ Wih1,
                             const float* __restrict__ Whh1,
                             unsigned short* __restrict__ ws16) {
  int f = blockIdx.x;        // 0..111  (48 layer0 frags, 64 layer1 frags)
  int l = threadIdx.x;       // 0..31
  int layer, tile, chunk;
  if (f < 48) { layer = 0; tile = f / 3; chunk = f % 3; }
  else        { layer = 1; tile = (f - 48) / 4; chunk = (f - 48) % 4; }
  int n  = tile * 16 + (l & 15);
  int kb = chunk * 32 + ((l >= 16) ? 8 : 0);
  unsigned short* dst = ws16 + ((size_t)(f * 32 + l)) * 16;
  for (int i = 0; i < 16; ++i) {
    int K = kb + (i & 7) + ((i & 8) << 1);
    float v = 0.0f;
    if (layer == 0) {
      if (K < 4)       v = Wih0[n * 4 + K];
      else if (K < 68) v = Whh0[n * 64 + (K - 4)];
    } else {
      if (K < 64)      v = Wih1[n * 64 + K];
      else             v = Whh1[n * 64 + (K - 64)];
    }
    dst[i] = f2bf(v);
  }
}

__global__ void prep_bias(const float* __restrict__ bi0, const float* __restrict__ bh0,
                          const float* __restrict__ bi1, const float* __restrict__ bh1,
                          float* __restrict__ biasws) {
  int n = threadIdx.x;       // 0..255
  biasws[n]       = bi0[n] + bh0[n];
  biasws[256 + n] = bi1[n] + bh1[n];
}

// ---------------------------------------------------------------------------
// Fused 2-layer LSTM + LayerNorm + FC.
// 128 threads = 4 waves; each wave owns 16 batch rows, scans t = 0..99.
// Dynamic LDS (bytes):
//   [0, 114688)      : weight fragments (112 frags * 32 lanes * 32B), TDM-loaded
//   114688 + w*3072  : a0frag — layer0 A staging [3 chunks][32 lanes][16 bf16]
//                      = [x_t(K0..3) | h0(K4..67) | 0]
//   126976 + w*4096  : a1frag — layer1 A staging [4 chunks][32][16]
//                      = [h0_out(K0..63) | h1(K64..127)]
//   143360 + w*4096  : hfin[16][64] f32 (last-step h1 for LN/FC)
// Total 159744 B -> 2 workgroups per 320KB WGP.
// ---------------------------------------------------------------------------
__launch_bounds__(128, 1)
__global__ void lstm_fused(const float* __restrict__ x,
                           const unsigned int* __restrict__ wfrag_g,
                           const float* __restrict__ bias_g,
                           const float* __restrict__ gamma,
                           const float* __restrict__ beta,
                           const float* __restrict__ Wfc,
                           const float* __restrict__ bfc,
                           float* __restrict__ out) {
  extern __shared__ __align__(16) unsigned char smem[];
  unsigned int* wfrag = (unsigned int*)smem;

  const int tid  = threadIdx.x;
  const int w    = tid >> 5;
  const int lane = tid & 31;

#if __has_builtin(__builtin_amdgcn_tensor_load_to_lds) && \
    __has_builtin(__builtin_amdgcn_s_wait_tensorcnt)
  // One-shot Tensor Data Mover DMA: 114688 B of pre-swizzled weight fragments
  // global -> LDS offset 0. D# per CDNA5 ISA ch.8: 2-D tensor of 14336 x 1
  // 8-byte elements, tile == tensor (contiguous). Issued by wave 0 only;
  // TDM ignores EXEC, completion tracked with TENSORcnt.
  if (w == 0) {
    unsigned long long ga = (unsigned long long)wfrag_g;
    v4u g0;
    g0[0] = 1u;                                   // count=1 valid descriptor
    g0[1] = 0u;                                   // lds_addr = 0 (WG LDS base)
    g0[2] = (unsigned int)ga;                     // global_addr[31:0]
    g0[3] = (unsigned int)(ga >> 32) | 0x80000000u; // global_addr[56:32] | type=2
    v8i g1;
    g1[0] = 0x00030000;   // workgroup_mask=0, data_size=3 (8B)
    g1[1] = 0x38000000;   // tensor_dim0 = 14336 (low16 at bits 63:48)
    g1[2] = 0x00010000;   // tensor_dim1 = 1
    g1[3] = 0x38000000;   // tile_dim0 = 14336
    g1[4] = 0x00000001;   // tile_dim1 = 1, tile_dim2 = 0
    g1[5] = 0x00003800;   // tensor_dim0_stride = 14336 (low32)
    g1[6] = 0x38000000;   // tensor_dim1_stride = 14336 (low16)
    g1[7] = 0;
    v4i gz  = {0, 0, 0, 0};                       // groups 2/3: dims unused
    v8i gz8 = {0, 0, 0, 0, 0, 0, 0, 0};           // extra group (clang-23 form)
    __builtin_amdgcn_tensor_load_to_lds(g0, g1, gz, gz, gz8, 0);
    __builtin_amdgcn_s_wait_tensorcnt(0);
  }
#else
  {   // fallback: cooperative vector-load copy (7168 uint4 across 128 threads)
    uint4* d = (uint4*)wfrag;
    const uint4* s = (const uint4*)wfrag_g;
    for (int i = tid; i < 7168; i += 128) d[i] = s[i];
  }
#endif

  unsigned short* a0frag = (unsigned short*)(smem + 114688 + w * 3072);
  unsigned short* a1frag = (unsigned short*)(smem + 126976 + w * 4096);
  float*          hfin   = (float*)(smem + 143360 + w * 4096);

  {   // zero A staging (h0 = h1 = 0; also the permanent K=68..95 zero pad)
    unsigned int* z0 = (unsigned int*)a0frag;
    for (int i = lane; i < 768; i += 32) z0[i] = 0u;
    unsigned int* z1 = (unsigned int*)a1frag;
    for (int i = lane; i < 1024; i += 32) z1[i] = 0u;
  }
  __syncthreads();

  const int nC     = lane & 15;            // C/D column within tile
  const int mCbase = (lane >> 4) << 3;     // C/D rows are M = r + mCbase

  // pre-scaled fused biases: sigmoid gates get 0.5*b, tanh gate gets b
  float bias0r[16], bias1r[16];
#pragma unroll
  for (int tle = 0; tle < 16; ++tle) {
    float s = ((tle >> 2) == 2) ? 1.0f : 0.5f;   // tiles 8..11 = g (tanh) gate
    bias0r[tle] = s * bias_g[tle * 16 + nC];
    bias1r[tle] = s * bias_g[256 + tle * 16 + nC];
  }

  // per-j scatter bases into A staging (invert the A-fragment layout)
  int w0base[4], w1ibase[4], w1hbase[4];
#pragma unroll
  for (int j = 0; j < 4; ++j) {
    int n = j * 16 + nC;
    w0base[j]  = fragIdx(4 + n);     // h0 into layer0 A (after x columns)
    w1ibase[j] = fragIdx(n);         // h0_out into layer1 A first half
    w1hbase[j] = fragIdx(64 + n);    // h1 into layer1 A second half
  }

  float c0st[32], c1st[32];
#pragma unroll
  for (int i = 0; i < 32; ++i) { c0st[i] = 0.0f; c1st[i] = 0.0f; }

  const int b0 = blockIdx.x * 64 + w * 16;
  const int xm = lane >> 1;                // 2 lanes per batch row for x load
  const int xd = (lane & 1) * 2;
  const float* xptr = x + ((size_t)(b0 + xm) * 100) * 4 + xd;
  unsigned int* a0d = (unsigned int*)a0frag;
  const int xslot = xm * 8 + (xd >> 1);    // chunk0, lane xm, halves xd..xd+1

  union AB { v16bf v; uint4 q[2]; };

  for (int t = 0; t < 100; ++t) {
    // --- stage x_t (bf16) directly into layer0 A-fragment chunk0 ---
    float xv0 = xptr[0], xv1 = xptr[1];
    a0d[xslot] = (unsigned int)f2bf_n(xv0) | ((unsigned int)f2bf_n(xv1) << 16);
    xptr += 4;
    asm volatile("s_wait_dscnt 0" ::: "memory");

    // --- layer0 A fragments: 2x b128 per chunk ---
    AB A0c[3];
#pragma unroll
    for (int c = 0; c < 3; ++c) {
      const uint4* p = (const uint4*)(a0frag + ((c * 32 + lane) << 4));
      A0c[c].q[0] = p[0]; A0c[c].q[1] = p[1];
    }

    // --- layer0: per 16-wide hidden column group j, gates i/f/g/o ---
#pragma unroll
    for (int j = 0; j < 4; ++j) {
      v8f accv[4];
#pragma unroll
      for (int g = 0; g < 4; ++g) {
        v8f acc = {};                       // inline SRC2 = 0
#pragma unroll
        for (int c = 0; c < 3; ++c) {
          int f = (g * 4 + j) * 3 + c;
          AB bf;
          const uint4* bp = (const uint4*)(wfrag + (f * 32 + lane) * 8);
          bf.q[0] = bp[0]; bf.q[1] = bp[1];
          acc = __builtin_amdgcn_wmma_f32_16x16x32_bf16(
              false, A0c[c].v, false, bf.v, (short)0, acc, false, false);
        }
        accv[g] = acc;
      }
#pragma unroll
      for (int r = 0; r < 8; ++r) {
        float gi = sgate(accv[0][r], bias0r[j]);
        float gf = sgate(accv[1][r], bias0r[4 + j]);
        float gg = hw_tanh(accv[2][r] + bias0r[8 + j]);
        float go = sgate(accv[3][r], bias0r[12 + j]);
        float cc = __builtin_fmaf(gf, c0st[j * 8 + r], gi * gg);
        c0st[j * 8 + r] = cc;
        float h = go * hw_tanh(cc);
        unsigned short hb = f2bf_n(h);
        int mr16 = (mCbase + r) << 4;
        a0frag[w0base[j] + mr16]  = hb;    // h0 state for next step
        a1frag[w1ibase[j] + mr16] = hb;    // input to layer1 this step
      }
    }
    asm volatile("s_wait_dscnt 0" ::: "memory");

    // --- layer1 A fragments ---
    AB A1c[4];
#pragma unroll
    for (int c = 0; c < 4; ++c) {
      const uint4* p = (const uint4*)(a1frag + ((c * 32 + lane) << 4));
      A1c[c].q[0] = p[0]; A1c[c].q[1] = p[1];
    }

    // --- layer1 ---
    bool last = (t == 99);
#pragma unroll
    for (int j = 0; j < 4; ++j) {
      v8f accv[4];
#pragma unroll
      for (int g = 0; g < 4; ++g) {
        v8f acc = {};
#pragma unroll
        for (int c = 0; c < 4; ++c) {
          int f = 48 + (g * 4 + j) * 4 + c;
          AB bf;
          const uint4* bp = (const uint4*)(wfrag + (f * 32 + lane) * 8);
          bf.q[0] = bp[0]; bf.q[1] = bp[1];
          acc = __builtin_amdgcn_wmma_f32_16x16x32_bf16(
              false, A1c[c].v, false, bf.v, (short)0, acc, false, false);
        }
        accv[g] = acc;
      }
#pragma unroll
      for (int r = 0; r < 8; ++r) {
        float gi = sgate(accv[0][r], bias1r[j]);
        float gf = sgate(accv[1][r], bias1r[4 + j]);
        float gg = hw_tanh(accv[2][r] + bias1r[8 + j]);
        float go = sgate(accv[3][r], bias1r[12 + j]);
        float cc = __builtin_fmaf(gf, c1st[j * 8 + r], gi * gg);
        c1st[j * 8 + r] = cc;
        float h = go * hw_tanh(cc);
        int mr = mCbase + r;
        a1frag[w1hbase[j] + (mr << 4)] = f2bf_n(h);
        if (last) hfin[mr * 64 + j * 16 + nC] = h;
      }
    }
    asm volatile("s_wait_dscnt 0" ::: "memory");
  }

  // --- LayerNorm + FC on last h1; one lane per batch row ---
  if (lane < 16) {
    const float* hr = hfin + lane * 64;
    float s = 0.0f;
    for (int k = 0; k < 64; ++k) s += hr[k];
    float mu = s * (1.0f / 64.0f);
    float v = 0.0f;
    for (int k = 0; k < 64; ++k) { float d = hr[k] - mu; v += d * d; }
    v *= (1.0f / 64.0f);
    float rinv = __frsqrt_rn(v + 1e-5f);
    float o0 = bfc[0], o1 = bfc[1];
    for (int k = 0; k < 64; ++k) {
      float nmv = (hr[k] - mu) * rinv * gamma[k] + beta[k];
      o0 += nmv * Wfc[k];
      o1 += nmv * Wfc[64 + k];
    }
    int b = b0 + lane;
    out[b * 2 + 0] = o0;
    out[b * 2 + 1] = o1;
  }
}

extern "C" void kernel_launch(void* const* d_in, const int* in_sizes, int n_in,
                              void* d_out, int out_size, void* d_ws, size_t ws_size,
                              hipStream_t stream) {
  const float* x    = (const float*)d_in[0];
  const float* Wih0 = (const float*)d_in[1];
  const float* Whh0 = (const float*)d_in[2];
  const float* bih0 = (const float*)d_in[3];
  const float* bhh0 = (const float*)d_in[4];
  const float* Wih1 = (const float*)d_in[5];
  const float* Whh1 = (const float*)d_in[6];
  const float* bih1 = (const float*)d_in[7];
  const float* bhh1 = (const float*)d_in[8];
  const float* gamma= (const float*)d_in[9];
  const float* beta = (const float*)d_in[10];
  const float* Wfc  = (const float*)d_in[11];
  const float* bfc  = (const float*)d_in[12];
  float* out = (float*)d_out;

  unsigned short* ws16 = (unsigned short*)d_ws;                 // 114688 B frag data
  float* biasws = (float*)((char*)d_ws + 114688);               // 512 f32 fused biases

  prep_weights<<<112, 32, 0, stream>>>(Wih0, Whh0, Wih1, Whh1, ws16);
  prep_bias<<<1, 256, 0, stream>>>(bih0, bhh0, bih1, bhh1, biasws);

  // 16384 rows / 64 rows per block = 256 blocks, 128 threads (4 wave32)
  lstm_fused<<<256, 128, 159744, stream>>>(x, (const unsigned int*)d_ws, biasws,
                                           gamma, beta, Wfc, bfc, out);
}
